// TransformerBlock_62423054680192
// MI455X (gfx1250) — compile-verified, matrix operand order checked by reading, not run on hardware
//
#include <hip/hip_runtime.h>
#include <hip/hip_bf16.h>

#define Bn 128
#define Cn 512
#define Sn 1024          // H*W
#define MT 64            // tokens per workgroup
#define STRA 520         // bf16 stride for A/H tiles (pad vs bank conflicts)
#define STRO 68          // f32 stride for out staging

typedef __attribute__((ext_vector_type(16))) __bf16 v16bf;
typedef __attribute__((ext_vector_type(8)))  __bf16 v8bf;
typedef __attribute__((ext_vector_type(8)))  float  v8f;

// LDS layout (dynamic):
//   sA : [MT][STRA] bf16   = 66560 B
//   sH : [MT][STRA] bf16   = 66560 B
//   sO : [Cn][STRO] f32    = 139264 B
#define OFF_A 0
#define OFF_H 66560
#define OFF_O 133120
#define LDS_BYTES (133120 + 139264)

// ---------------- pass 1: per-sample mean / rsigma ----------------
__global__ void ln_stats_kernel(const float* __restrict__ x, float* __restrict__ stats) {
    const int b = blockIdx.x, tid = threadIdx.x;
    const float4* xb = (const float4*)(x + (size_t)b * Cn * Sn);
    float s = 0.f, s2 = 0.f;
    for (int i = tid; i < (Cn * Sn) / 4; i += 256) {
        float4 v = xb[i];
        s  += v.x + v.y + v.z + v.w;
        s2 += v.x * v.x + v.y * v.y + v.z * v.z + v.w * v.w;
    }
    __shared__ float rs[256], rq[256];
    rs[tid] = s; rq[tid] = s2;
    __syncthreads();
    for (int off = 128; off > 0; off >>= 1) {
        if (tid < off) { rs[tid] += rs[tid + off]; rq[tid] += rq[tid + off]; }
        __syncthreads();
    }
    if (tid == 0) {
        const float n  = (float)(Cn * Sn);
        const float mu = rs[0] / n;
        const float var = rq[0] / n - mu * mu;
        stats[2 * b + 0] = mu;
        stats[2 * b + 1] = rsqrtf(var + 1e-5f);
    }
}

// ---------------- weight fp32 -> bf16 ----------------
__global__ void cvt_bf16_kernel(const float* __restrict__ w, __bf16* __restrict__ o, int n) {
    int i = blockIdx.x * 256 + threadIdx.x;
    if (i < n) o[i] = (__bf16)w[i];
}

// Load one 16-half fragment chunk pair per the gfx1250 16-bit A/B layout:
// lanes 0-15 take K chunks {k..k+7, k+16..k+23}, lanes 16-31 {k+8..k+15, k+24..k+31}.
// `p` is already offset by the per-lane row/col base and the lane-half select.
__device__ __forceinline__ v16bf ldfrag(const __bf16* __restrict__ p, int k) {
    v8bf lo = *(const v8bf*)(p + k);
    v8bf hi = *(const v8bf*)(p + k + 16);
    return __builtin_shufflevector(lo, hi, 0,1,2,3,4,5,6,7,8,9,10,11,12,13,14,15);
}

#define WMMA_BF16(A, Bm, C) \
    __builtin_amdgcn_wmma_f32_16x16x32_bf16(false, (A), false, (Bm), (short)0, (C), false, false)

// One wave computes a 16-row x 256-col slab of the 64x512 output, K = 512.
// 4-tile n-blocking (1 A fragment -> 4 independent WMMA chains) + ping-pong
// double buffering with no rotation copies: each fragment variable is defined
// exactly once per iteration by a load, so loads target the WMMA operand
// registers directly.
template <bool RELU>
__device__ __forceinline__ void gemm_slab(const __bf16* __restrict__ aLds,
                                          const __bf16* __restrict__ wG,
                                          const float*  __restrict__ bias,
                                          __bf16* __restrict__ sHout,
                                          float*  __restrict__ sOout,
                                          int mrow, int srow, int nh, int lane) {
    const int sel = (lane < 16) ? 0 : 8;
    const int cl  = lane & 15;
    const __bf16* aRow = aLds + mrow * STRA + sel;

    for (int g = 0; g < 4; ++g) {
        const int n0 = nh * 256 + g * 64;
        const __bf16* wp0 = wG + (size_t)(n0 + cl) * Cn + sel;
        const __bf16* wp1 = wp0 + 16 * Cn;
        const __bf16* wp2 = wp1 + 16 * Cn;
        const __bf16* wp3 = wp2 + 16 * Cn;
        if (g < 3) {                       // pull next group's weight rows toward L2/L0
            __builtin_prefetch(wp3 + 16 * Cn, 0, 1);
            __builtin_prefetch(wp3 + 32 * Cn, 0, 1);
        }

        v8f acc0 = {}, acc1 = {}, acc2 = {}, acc3 = {};

        // prologue: fragments for kb = 0 (buffer A)
        v16bf a0 = ldfrag(aRow, 0);
        v16bf p0 = ldfrag(wp0, 0), p1 = ldfrag(wp1, 0);
        v16bf p2 = ldfrag(wp2, 0), p3 = ldfrag(wp3, 0);

        #pragma unroll 1
        for (int kb = 0; kb < 14; kb += 2) {
            const int k1 = (kb + 1) * 32;
            const int k2 = (kb + 2) * 32;
            // buffer B <- kb+1
            v16bf a1 = ldfrag(aRow, k1);
            v16bf q0 = ldfrag(wp0, k1), q1 = ldfrag(wp1, k1);
            v16bf q2 = ldfrag(wp2, k1), q3 = ldfrag(wp3, k1);
            // math on buffer A (kb)
            acc0 = WMMA_BF16(a0, p0, acc0);
            acc1 = WMMA_BF16(a0, p1, acc1);
            acc2 = WMMA_BF16(a0, p2, acc2);
            acc3 = WMMA_BF16(a0, p3, acc3);
            // buffer A <- kb+2
            a0 = ldfrag(aRow, k2);
            p0 = ldfrag(wp0, k2); p1 = ldfrag(wp1, k2);
            p2 = ldfrag(wp2, k2); p3 = ldfrag(wp3, k2);
            // math on buffer B (kb+1)
            acc0 = WMMA_BF16(a1, q0, acc0);
            acc1 = WMMA_BF16(a1, q1, acc1);
            acc2 = WMMA_BF16(a1, q2, acc2);
            acc3 = WMMA_BF16(a1, q3, acc3);
        }
        {   // peeled kb = 14, 15 (no conditional loads in the hot loop)
            const int k1 = 15 * 32;
            v16bf a1 = ldfrag(aRow, k1);
            v16bf q0 = ldfrag(wp0, k1), q1 = ldfrag(wp1, k1);
            v16bf q2 = ldfrag(wp2, k1), q3 = ldfrag(wp3, k1);
            acc0 = WMMA_BF16(a0, p0, acc0);
            acc1 = WMMA_BF16(a0, p1, acc1);
            acc2 = WMMA_BF16(a0, p2, acc2);
            acc3 = WMMA_BF16(a0, p3, acc3);
            acc0 = WMMA_BF16(a1, q0, acc0);
            acc1 = WMMA_BF16(a1, q1, acc1);
            acc2 = WMMA_BF16(a1, q2, acc2);
            acc3 = WMMA_BF16(a1, q3, acc3);
        }

        v8f accs[4] = {acc0, acc1, acc2, acc3};
        #pragma unroll
        for (int t = 0; t < 4; ++t) {
            const int col = n0 + t * 16 + cl;
            const float bv = bias[col];
            #pragma unroll
            for (int r = 0; r < 8; ++r) {
                float v = accs[t][r] + bv;
                if (RELU) {
                    v = v > 0.f ? v : 0.f;
                    sHout[(srow + r) * STRA + col] = (__bf16)v;
                } else {
                    sOout[col * STRO + (srow + r)] = v;
                }
            }
        }
    }
}

// ---------------- fused LN-apply + MLP (GEMM1+ReLU+GEMM2) ----------------
__global__ void fused_mlp_kernel(const float* __restrict__ x,
                                 const float* __restrict__ ln_w,
                                 const float* __restrict__ ln_b,
                                 const __bf16* __restrict__ w1b,
                                 const float* __restrict__ b1,
                                 const __bf16* __restrict__ w2b,
                                 const float* __restrict__ b2,
                                 const float* __restrict__ stats,
                                 float* __restrict__ out) {
    extern __shared__ char smem[];
    __bf16* sA = (__bf16*)(smem + OFF_A);
    __bf16* sH = (__bf16*)(smem + OFF_H);
    float*  sO = (float*) (smem + OFF_O);

    const int tile = blockIdx.x;            // token tile: 0..15
    const int b    = blockIdx.y;            // batch
    const int s0   = tile * MT;
    const int tid  = threadIdx.x;
    const int wave = tid >> 5;
    const int lane = tid & 31;

    const float mu = stats[2 * b + 0];
    const float rs = stats[2 * b + 1];

    // ---- stage normalized bf16 A tile: sA[s_local][c] ----
    const float* xb = x + (size_t)b * Cn * Sn;
    #pragma unroll 4
    for (int i = 0; i < (MT * Cn) / 256; ++i) {
        int flat = tid + i * 256;
        int c    = flat >> 6;               // /MT
        int si   = flat & (MT - 1);
        size_t g = (size_t)c * Sn + s0 + si;
        float xn = (xb[g] - mu) * rs * ln_w[g] + ln_b[g];
        sA[si * STRA + c] = (__bf16)xn;
    }
    __syncthreads();

    // wave -> 16 rows x 256 cols of the 64x512 slab
    const int m0   = (wave & 3) * 16;
    const int nh   = (wave >> 2) & 1;
    const int mrow = m0 + (lane & 15);
    const int srow = m0 + ((lane & 16) ? 8 : 0);

    // GEMM1: hdn = relu(A @ w1^T + b1) -> sH (bf16)
    gemm_slab<true>(sA, w1b, b1, sH, sO, mrow, srow, nh, lane);
    __syncthreads();

    // GEMM2: out = hdn @ w2^T + b2 -> sO (f32, transposed [e][s])
    gemm_slab<false>(sH, w2b, b2, sH, sO, mrow, srow, nh, lane);
    __syncthreads();

    // ---- coalesced writeback: out[b][e][s0..s0+63] ----
    float* ob = out + (size_t)b * Cn * Sn;
    #pragma unroll 4
    for (int i = 0; i < (Cn * MT / 4) / 256; ++i) {
        int flat = tid + i * 256;
        int e  = flat >> 4;
        int sv = (flat & 15) << 2;
        float4 v = *(const float4*)&sO[e * STRO + sv];
        *(float4*)&ob[(size_t)e * Sn + s0 + sv] = v;
    }
}

extern "C" void kernel_launch(void* const* d_in, const int* in_sizes, int n_in,
                              void* d_out, int out_size, void* d_ws, size_t ws_size,
                              hipStream_t stream) {
    const float* x    = (const float*)d_in[0];
    const float* ln_w = (const float*)d_in[1];
    const float* ln_b = (const float*)d_in[2];
    const float* w1   = (const float*)d_in[3];
    const float* b1   = (const float*)d_in[4];
    const float* w2   = (const float*)d_in[5];
    const float* b2   = (const float*)d_in[6];
    float* out = (float*)d_out;

    // workspace layout: [stats: 2*B f32 = 1024 B][w1 bf16: 512 KB][w2 bf16: 512 KB]
    float*  stats = (float*)d_ws;
    __bf16* w1b   = (__bf16*)((char*)d_ws + 1024);
    __bf16* w2b   = (__bf16*)((char*)d_ws + 1024 + Cn * Cn * 2);

    hipFuncSetAttribute((const void*)fused_mlp_kernel,
                        hipFuncAttributeMaxDynamicSharedMemorySize, LDS_BYTES);

    ln_stats_kernel<<<Bn, 256, 0, stream>>>(x, stats);
    cvt_bf16_kernel<<<(Cn * Cn + 255) / 256, 256, 0, stream>>>(w1, w1b, Cn * Cn);
    cvt_bf16_kernel<<<(Cn * Cn + 255) / 256, 256, 0, stream>>>(w2, w2b, Cn * Cn);

    dim3 grid(Sn / MT, Bn);
    fused_mlp_kernel<<<grid, 256, LDS_BYTES, stream>>>(
        x, ln_w, ln_b, w1b, b1, w2b, b2, stats, out);
}
